// Attention_46566035423271
// MI455X (gfx1250) — matrix operand order repaired and checked by reading, not compiled, not run on hardware
//
#include <hip/hip_runtime.h>

typedef __attribute__((ext_vector_type(16))) _Float16 v16h;
typedef __attribute__((ext_vector_type(8)))  _Float16 v8h;
typedef __attribute__((ext_vector_type(8)))  float    v8f;

#define SCALE_ 0.25f  // HD^-0.5, HD=16

__device__ __forceinline__ v8f wmma_f16(v16h a, v16h b, v8f c) {
  return __builtin_amdgcn_wmma_f32_16x16x32_f16(false, a, false, b, (short)0, c, false, false);
}

// A tile loader: row-major f16 [16][ld], K-chunk starting at k0 (32 wide).
__device__ __forceinline__ v16h load_a16(const _Float16* A, int ld, int k0, int lane) {
  int m = lane & 15, kb = (lane >> 4) * 16;
  const _Float16* p = A + m * ld + k0 + kb;
  v16h r;
#pragma unroll
  for (int j = 0; j < 8; ++j) { r[j] = p[j]; r[8 + j] = p[8 + j]; }
  return r;
}

// B tile loader from fp32 weight W (row-major [n][K]); B[k][n] = W[n][k].
__device__ __forceinline__ v16h load_b_w32(const float* Wrow, int ldw, int k0, int lane) {
  int n = lane & 15, kb = k0 + (lane >> 4) * 16;
  const float* p = Wrow + n * ldw + kb;
  v16h r;
#pragma unroll
  for (int j = 0; j < 16; ++j) r[j] = (_Float16)p[j];
  return r;
}

__global__ void k_zero(float* p) {
  p[blockIdx.x * blockDim.x + threadIdx.x] = 0.0f;
}

// LDS-tiled transpose: x (NCHW fp32) -> xt16[p][c] and xkt16[(a*256+w)][b].
// blocks 0..1023:   xt, 64x64 tile at p0 = bid*64       (row stride 65536)
// blocks 1024..2047: xkt, a = (bid-1024)>>2, w0 = ((bid-1024)&3)*64 (row stride 256)
__global__ void k_transpose(const float* __restrict__ x, _Float16* __restrict__ xt16,
                            _Float16* __restrict__ xkt16) {
  __shared__ _Float16 tile[64 * 65];
  int bid = blockIdx.x;
  const float* src; _Float16* dst; int rstride;
  if (bid < 1024) {
    src = x + bid * 64;
    rstride = 65536;
    dst = xt16 + bid * 64 * 64;
  } else {
    int a = (bid - 1024) >> 2, wt = (bid - 1024) & 3;
    src = x + a * 16384 + wt * 64;
    rstride = 256;
    dst = xkt16 + (a * 256 + wt * 64) * 64;
  }
  int t = threadIdx.x;
#pragma unroll
  for (int i = 0; i < 16; ++i) {
    int idx = i * 256 + t;
    int c = idx >> 6, pp = idx & 63;             // coalesced fp32 reads over pp
    tile[c * 65 + pp] = (_Float16)src[c * rstride + pp];
  }
  __syncthreads();
#pragma unroll
  for (int i = 0; i < 2; ++i) {
    int idx = i * 256 + t;                        // 512 chunks of 8 halves
    int pp = idx >> 3, c0 = (idx & 7) * 8;
    v8h h8;
#pragma unroll
    for (int j = 0; j < 8; ++j) h8[j] = tile[(c0 + j) * 65 + pp];
    *reinterpret_cast<v8h*>(dst + pp * 64 + c0) = h8;   // b128 coalesced write
  }
}

// Fused GEMM: q1 (N 0..31), q2 (N 32..63 of q1q2) staged via LDS for coalesced
// b128 stores; lepe-linear (+bias) written NCHW with packed b128 per lane.
__global__ void k_qlepe(const _Float16* __restrict__ xt16, const float* __restrict__ w_q1,
                        const float* __restrict__ w_q2, const float* __restrict__ w_lepe,
                        const float* __restrict__ b_lepe, _Float16* __restrict__ q1q2,
                        _Float16* __restrict__ lepe16) {
  __shared__ _Float16 Cs[8][16][64];
  int wave = threadIdx.x >> 5, lane = threadIdx.x & 31;
  int wv = blockIdx.x * 8 + wave;                 // M tile (4096 total)
  const _Float16* A = xt16 + wv * 16 * 64;
  v16h a0 = load_a16(A, 64, 0, lane);
  v16h a1 = load_a16(A, 64, 32, lane);
  int row0 = wv * 16;
  // q1 | q2 -> LDS tile
  for (int nt = 0; nt < 4; ++nt) {
    const float* Wm = (nt < 2) ? (w_q1 + nt * 16 * 64) : (w_q2 + (nt - 2) * 16 * 64);
    v16h b0 = load_b_w32(Wm, 64, 0, lane);
    v16h b1 = load_b_w32(Wm, 64, 32, lane);
    v8f c = {0,0,0,0,0,0,0,0};
    c = wmma_f16(a0, b0, c);
    c = wmma_f16(a1, b1, c);
    int n = lane & 15;
#pragma unroll
    for (int r = 0; r < 8; ++r)
      Cs[wave][r + 8 * (lane >> 4)][nt * 16 + n] = (_Float16)c[r];
  }
  // lepe-linear -> NCHW packed stores (8 consecutive p per lane)
  for (int nt = 0; nt < 4; ++nt) {
    v16h b0 = load_b_w32(w_lepe + nt * 16 * 64, 64, 0, lane);
    v16h b1 = load_b_w32(w_lepe + nt * 16 * 64, 64, 32, lane);
    v8f c = {0,0,0,0,0,0,0,0};
    c = wmma_f16(a0, b0, c);
    c = wmma_f16(a1, b1, c);
    int ch = nt * 16 + (lane & 15);
    float bias = b_lepe[ch];
    v8h h8;
#pragma unroll
    for (int r = 0; r < 8; ++r) h8[r] = (_Float16)(c[r] + bias);
    *reinterpret_cast<v8h*>(lepe16 + ch * 65536 + row0 + 8 * (lane >> 4)) = h8;
  }
  __syncthreads();
  // coalesced q1q2 store: 16 rows x 128B contiguous
#pragma unroll
  for (int i = 0; i < 4; ++i) {
    int idx = i * 32 + lane;
    int rr = idx >> 3, c0 = (idx & 7) * 8;
    v8h h8;
#pragma unroll
    for (int j = 0; j < 8; ++j) h8[j] = Cs[wave][rr][c0 + j];
    *reinterpret_cast<v8h*>(q1q2 + (row0 + rr) * 64 + c0) = h8;
  }
}

// kv2 = xkt @ w_kv2^T (scrambled token view), LDS-staged coalesced f16 out [t][64].
__global__ void k_kv2(const _Float16* __restrict__ xkt16, const float* __restrict__ w_kv2,
                      _Float16* __restrict__ kv2b) {
  __shared__ _Float16 Cs[8][16][64];
  int wave = threadIdx.x >> 5, lane = threadIdx.x & 31;
  int wv = blockIdx.x * 8 + wave;
  const _Float16* A = xkt16 + wv * 16 * 64;
  v16h a0 = load_a16(A, 64, 0, lane);
  v16h a1 = load_a16(A, 64, 32, lane);
  int row0 = wv * 16;
  for (int nt = 0; nt < 4; ++nt) {
    v16h b0 = load_b_w32(w_kv2 + nt * 16 * 64, 64, 0, lane);
    v16h b1 = load_b_w32(w_kv2 + nt * 16 * 64, 64, 32, lane);
    v8f c = {0,0,0,0,0,0,0,0};
    c = wmma_f16(a0, b0, c);
    c = wmma_f16(a1, b1, c);
    int n = lane & 15;
#pragma unroll
    for (int r = 0; r < 8; ++r)
      Cs[wave][r + 8 * (lane >> 4)][nt * 16 + n] = (_Float16)c[r];
  }
  __syncthreads();
#pragma unroll
  for (int i = 0; i < 4; ++i) {
    int idx = i * 32 + lane;
    int rr = idx >> 3, c0 = (idx & 7) * 8;
    v8h h8;
#pragma unroll
    for (int j = 0; j < 8; ++j) h8[j] = Cs[wave][rr][c0 + j];
    *reinterpret_cast<v8h*>(kv2b + (row0 + rr) * 64 + c0) = h8;
  }
}

// Depthwise 3x3 SAME conv on lepe16 (NCHW) + bias -> lepc16.
__global__ void k_lepeconv(const _Float16* __restrict__ lepe16, const float* __restrict__ k_lepe,
                           const float* __restrict__ kb_lepe, _Float16* __restrict__ lepc16) {
  int i = blockIdx.x * blockDim.x + threadIdx.x;  // c*65536 + p
  int c = i >> 16, p = i & 65535;
  int h = p >> 8, w = p & 255;
  const float* kw = k_lepe + c * 9;
  const _Float16* base = lepe16 + c * 65536;
  float acc = kb_lepe[c];
#pragma unroll
  for (int di = -1; di <= 1; ++di) {
    int hh = h + di;
    if (hh < 0 || hh > 255) continue;
#pragma unroll
    for (int dj = -1; dj <= 1; ++dj) {
      int ww = w + dj;
      if (ww < 0 || ww > 255) continue;
      acc += (float)base[hh * 256 + ww] * kw[(di + 1) * 3 + (dj + 1)];
    }
  }
  lepc16[i] = (_Float16)acc;
}

// SR conv as GEMM: M=1024 positions, N=64, K=4096 (ic,kh,kw). + b_sr.
__global__ void k_srconv(const float* __restrict__ x, const float* __restrict__ w_sr,
                         const float* __restrict__ b_sr, float* __restrict__ xs) {
  int wave = blockIdx.x * 8 + (threadIdx.x >> 5);  // 256 waves = 64 Mtiles x 4 Ntiles
  int lane = threadIdx.x & 31;
  int Mt = wave >> 2, Nt = wave & 3;
  int m = lane & 15;
  int pos = Mt * 16 + m, hs = pos >> 5, wsp = pos & 31;
  int oc = Nt * 16 + (lane & 15);
  v8f acc = {0,0,0,0,0,0,0,0};
  for (int kk = 0; kk < 128; ++kk) {
    int k0 = kk * 32;
    v16h a, b;
    int kbA = k0 + (lane >> 4) * 16;
#pragma unroll
    for (int g = 0; g < 2; ++g) {
      int kb = kbA + g * 8;                  // 8-wide run: fixed (ic,kh), kw 0..7
      int ic = kb >> 6, kh = (kb >> 3) & 7;
      const float* p = x + ic * 65536 + (hs * 8 + kh) * 256 + wsp * 8;
      __builtin_prefetch(p + 512, 0, 1);
#pragma unroll
      for (int j = 0; j < 8; ++j) a[g * 8 + j] = (_Float16)p[j];
    }
    int kbB = k0 + (lane >> 4) * 16;
    const float* pb = w_sr + oc * 4096 + kbB;
#pragma unroll
    for (int j = 0; j < 16; ++j) b[j] = (_Float16)pb[j];
    acc = wmma_f16(a, b, acc);
  }
  float bias = b_sr[oc];
#pragma unroll
  for (int r = 0; r < 8; ++r) {
    int mm = r + 8 * (lane >> 4);
    xs[(Mt * 16 + mm) * 64 + oc] = acc[r] + bias;
  }
}

// Single workgroup: ada-ILN stats + blend + GELU, then kv1 GEMM (M=1024,N=64,K=64).
// Writes k1s[hs][nh][j=ws][d] and v1sT[hs][nh][d][j=ws] in f16.
__global__ void __launch_bounds__(1024)
k_kv1norm(const float* __restrict__ xs, const float* __restrict__ rho,
          const float* __restrict__ gamma, const float* __restrict__ beta,
          const float* __restrict__ w_kv1, _Float16* __restrict__ xsn16,
          _Float16* __restrict__ k1s, _Float16* __restrict__ v1sT) {
  __shared__ float redS[64][17], redQ[64][17];
  __shared__ float chMu[64], chR[64];
  __shared__ float lnS[2];
  int t = threadIdx.x;
  int c = t & 63, seg = t >> 6;
  float s = 0.0f, q = 0.0f;
  for (int i = 0; i < 64; ++i) {
    float v = xs[(seg * 64 + i) * 64 + c];
    s += v; q += v * v;
  }
  redS[c][seg] = s; redQ[c][seg] = q;
  __syncthreads();
  if (t < 64) {
    float ts = 0.0f, tq = 0.0f;
    for (int i = 0; i < 16; ++i) { ts += redS[t][i]; tq += redQ[t][i]; }
    float mu = ts * (1.0f / 1024.0f);
    float var = tq * (1.0f / 1024.0f) - mu * mu;
    chMu[t] = mu; chR[t] = rsqrtf(var + 1e-5f);
    redS[t][16] = ts; redQ[t][16] = tq;
  }
  __syncthreads();
  if (t == 0) {
    float ts = 0.0f, tq = 0.0f;
    for (int i = 0; i < 64; ++i) { ts += redS[i][16]; tq += redQ[i][16]; }
    float mu = ts * (1.0f / 65536.0f);
    float var = tq * (1.0f / 65536.0f) - mu * mu;
    lnS[0] = mu; lnS[1] = rsqrtf(var + 1e-5f);
  }
  __syncthreads();
  float lmu = lnS[0], lr = lnS[1];
  for (int i = 0; i < 64; ++i) {
    int idx = t + i * 1024;
    int cc = idx & 63;
    float v = xs[idx];
    float xin = (v - chMu[cc]) * chR[cc];
    float xln = (v - lmu) * lr;
    float rr = rho[cc];
    float y = (rr * xin + (1.0f - rr) * xln) * gamma[cc] + beta[cc];
    xsn16[idx] = (_Float16)(0.5f * y * (1.0f + erff(y * 0.70710678118654752f)));
  }
  __threadfence();
  __syncthreads();
  int wave = t >> 5, lane = t & 31;
  for (int it = 0; it < 8; ++it) {
    int tile = wave * 8 + it;                 // 256 tiles over 32 waves
    int Mt = tile >> 2, Nt = tile & 3;
    const _Float16* A = xsn16 + Mt * 16 * 64;
    v16h a0 = load_a16(A, 64, 0, lane);
    v16h a1 = load_a16(A, 64, 32, lane);
    v16h b0 = load_b_w32(w_kv1 + Nt * 16 * 64, 64, 0, lane);
    v16h b1 = load_b_w32(w_kv1 + Nt * 16 * 64, 64, 32, lane);
    v8f cacc = {0,0,0,0,0,0,0,0};
    cacc = wmma_f16(a0, b0, cacc);
    cacc = wmma_f16(a1, b1, cacc);
    int col = Nt * 16 + (lane & 15);
#pragma unroll
    for (int r = 0; r < 8; ++r) {
      int mm = r + 8 * (lane >> 4);
      int pos = Mt * 16 + mm, hs = pos >> 5, wsp = pos & 31;
      _Float16 hv = (_Float16)cacc[r];
      if (col < 32) {
        int nh = col >> 4, d = col & 15;
        k1s[((hs * 2 + nh) * 32 + wsp) * 16 + d] = hv;
      } else {
        int c2 = col - 32, nh = c2 >> 4, d = c2 & 15;
        v1sT[((hs * 2 + nh) * 16 + d) * 32 + wsp] = hv;
      }
    }
  }
}

// Global branch row attention: per image row h, per head: S=Q@K^T (256x32 via WMMA,
// K=16 zero-padded to 32), softmax over 32 distinct keys (8x tiling folded into the
// normalizer + 1/8 mask factor), x1 = P@V via WMMA (16x16x32 exact).
__global__ void k_attn1(const _Float16* __restrict__ q1q2, const _Float16* __restrict__ k1s,
                        const _Float16* __restrict__ v1sT, float* __restrict__ x1,
                        float* __restrict__ gmask) {
  __shared__ float S[256][33];
  __shared__ _Float16 P16[256][32];
  __shared__ float colsum[32];
  int h = blockIdx.x, t = threadIdx.x;
  int wave = t >> 5, lane = t & 31;
  if (t < 32) colsum[t] = 0.0f;
  __syncthreads();
  for (int nh = 0; nh < 2; ++nh) {
    for (int mi = 0; mi < 2; ++mi) {               // logits
      int mt = wave * 2 + mi;
      int mQ = lane & 15, kb = 0;
      v16h a;
      const _Float16* pq = q1q2 + (h * 256 + mt * 16 + mQ) * 64 + nh * 16 + kb;
#pragma unroll
      for (int j = 0; j < 8; ++j) { a[j] = pq[j]; a[8 + j] = pq[8 + j]; }
      for (int ntt = 0; ntt < 2; ++ntt) {
        int ncol = lane & 15;
        v16h b;
        if (lane < 16) {
          const _Float16* pk = k1s + (((h & 31) * 2 + nh) * 32 + ntt * 16 + ncol) * 16;
#pragma unroll
          for (int j = 0; j < 16; ++j) b[j] = pk[j];
        } else {
#pragma unroll
          for (int j = 0; j < 16; ++j) b[j] = (_Float16)0.0f;
        }
        v8f cz = {0,0,0,0,0,0,0,0};
        cz = wmma_f16(a, b, cz);
#pragma unroll
        for (int r = 0; r < 8; ++r)
          S[mt * 16 + r + 8 * (lane >> 4)][ntt * 16 + ncol] = cz[r];
      }
    }
    __syncthreads();
    {                                              // softmax (thread = query)
      float sv[32], mx = -3.0e38f;
#pragma unroll
      for (int j = 0; j < 32; ++j) { sv[j] = S[t][j] * SCALE_; mx = fmaxf(mx, sv[j]); }
      float sum = 0.0f;
#pragma unroll
      for (int j = 0; j < 32; ++j) { sv[j] = __expf(sv[j] - mx); sum += sv[j]; }
      float inv = 1.0f / sum;
#pragma unroll
      for (int j = 0; j < 32; ++j) {
        float pp = sv[j] * inv;
        P16[t][j] = (_Float16)pp;
        atomicAdd(&colsum[j], pp);
      }
    }
    __syncthreads();
    for (int mi = 0; mi < 2; ++mi) {               // x1 = P @ V
      int mt = wave * 2 + mi;
      int mQ = lane & 15, kb = (lane >> 4) * 16;
      v16h a;
      const _Float16* pp = &P16[mt * 16 + mQ][kb];
#pragma unroll
      for (int j = 0; j < 8; ++j) { a[j] = pp[j]; a[8 + j] = pp[8 + j]; }
      int d = lane & 15, kbb = (lane >> 4) * 16;
      v16h b;
      const _Float16* pv = v1sT + (((h & 31) * 2 + nh) * 16 + d) * 32 + kbb;
#pragma unroll
      for (int j = 0; j < 16; ++j) b[j] = pv[j];
      v8f cz = {0,0,0,0,0,0,0,0};
      cz = wmma_f16(a, b, cz);
#pragma unroll
      for (int r = 0; r < 8; ++r)
        x1[(h * 256 + mt * 16 + r + 8 * (lane >> 4)) * 32 + nh * 16 + d] = cz[r];
    }
    __syncthreads();
  }
  gmask[h * 256 + t] = colsum[t & 31] * (1.0f / (2.0f * 256.0f * 8.0f));
}

// Local branch: 8x8 window row-attention (one thread per (window,row,query)).
__global__ void k_attn2(const _Float16* __restrict__ q1q2, const _Float16* __restrict__ kv2b,
                        float* __restrict__ wr, float* __restrict__ lm) {
  int t = blockIdx.x * blockDim.x + threadIdx.x;
  int win = t >> 6, r = (t >> 3) & 7, qq = t & 7;
  int np = win >> 10, bh = (win >> 5) & 31, bw = win & 31;
  int g = np * 256 + bh * 8 + r;                 // scrambled (h, head) split
  int hsrc = g >> 1, nh = g & 1;
  float qv[16];
  const _Float16* qp = q1q2 + (hsrc * 256 + bw * 8 + qq) * 64 + 32 + nh * 16;
#pragma unroll
  for (int d = 0; d < 16; ++d) qv[d] = (float)qp[d];
  float s[8], mx = -3.0e38f;
#pragma unroll
  for (int kk = 0; kk < 8; ++kk) {
    const _Float16* kp = kv2b + (hsrc * 256 + bw * 8 + kk) * 64 + nh * 16;
    float acc = 0.0f;
#pragma unroll
    for (int d = 0; d < 16; ++d) acc += qv[d] * (float)kp[d];
    s[kk] = acc * SCALE_;
    mx = fmaxf(mx, s[kk]);
  }
  float sum = 0.0f;
#pragma unroll
  for (int kk = 0; kk < 8; ++kk) { s[kk] = __expf(s[kk] - mx); sum += s[kk]; }
  float inv = 1.0f / sum;
  float out[16];
#pragma unroll
  for (int d = 0; d < 16; ++d) out[d] = 0.0f;
#pragma unroll
  for (int kk = 0; kk < 8; ++kk) {
    float pp = s[kk] * inv;
    const _Float16* vp = kv2b + (hsrc * 256 + bw * 8 + kk) * 64 + 32 + nh * 16;
#pragma unroll
    for (int d = 0; d < 16; ++d) out[d] += pp * (float)vp[d];
    atomicAdd(&lm[(bh * 8 + r) * 256 + bw * 8 + kk], pp * (1.0f / 16.0f));
  }
  float* wp = wr + ((bh * 8 + r) * 256 + bw * 8 + qq) * 32 + np * 16;
#pragma unroll
  for (int d = 0; d < 16; ++d) wp[d] = out[d];
}

// Final: gather concat(x1, scrambled x2) + scrambled lepe into LDS, WMMA @ w_proj + b_proj.
__global__ void k_final(const float* __restrict__ x1, const float* __restrict__ wr,
                        const _Float16* __restrict__ lepc16, const float* __restrict__ w_proj,
                        const float* __restrict__ b_proj, float* __restrict__ out) {
  __shared__ _Float16 As[8][16 * 64];
  int wave = threadIdx.x >> 5, lane = threadIdx.x & 31;
  int Mt = blockIdx.x * 8 + wave;
  int p0 = Mt * 16;
  for (int i = 0; i < 32; ++i) {
    int idx = lane + i * 32;
    int row = idx >> 6, cc = idx & 63;
    int p = p0 + row, h = p >> 8, w = p & 255;
    float v;
    if (cc < 32) v = x1[p * 32 + cc];
    else {
      int c2 = cc - 32;   // x2 scramble: wr[(h%8)*8192 + w*32 + c2][h/8]
      v = wr[((h & 7) * 8192 + w * 32 + c2) * 32 + (h >> 3)];
    }
    int t2 = w * 64 + cc; // lepe scramble
    float lep = (float)lepc16[(h >> 2) * 65536 + ((h & 3) * 64 + (t2 >> 8)) * 256 + (t2 & 255)];
    As[wave][row * 64 + cc] = (_Float16)(v + lep);
  }
  __syncthreads();
  v16h a0 = load_a16(&As[wave][0], 64, 0, lane);
  v16h a1 = load_a16(&As[wave][0], 64, 32, lane);
  for (int nt = 0; nt < 4; ++nt) {
    v16h b0 = load_b_w32(w_proj + nt * 16 * 64, 64, 0, lane);
    v16h b1 = load_b_w32(w_proj + nt * 16 * 64, 64, 32, lane);
    v8f c = {0,0,0,0,0,0,0,0};
    c = wmma_f16(a0, b0, c);
    c = wmma_f16(a1, b1, c);
    int col = nt * 16 + (lane & 15);
    float bias = b_proj[col];
#pragma unroll
    for (int r = 0; r < 8; ++r)
      out[(p0 + r + 8 * (lane >> 4)) * 64 + col] = c[r] + bias;
  }
}

__global__ void k_mask(const float* __restrict__ lm, const float* __restrict__ gmask,
                       float* __restrict__ outm) {
  int i = blockIdx.x * blockDim.x + threadIdx.x;
  float v = lm[i] + gmask[i];
  outm[i] = v;                                          // mask_1
  outm[65536 + ((i & 255) << 8) + (i >> 8)] = v;        // mask_2 (transposed)
}

extern "C" void kernel_launch(void* const* d_in, const int* in_sizes, int n_in,
                              void* d_out, int out_size, void* d_ws, size_t ws_size,
                              hipStream_t stream) {
  (void)in_sizes; (void)n_in; (void)out_size; (void)ws_size;
  const float* x      = (const float*)d_in[0];
  const float* gamma  = (const float*)d_in[1];
  const float* beta   = (const float*)d_in[2];
  const float* rho    = (const float*)d_in[3];
  const float* w_lepe = (const float*)d_in[4];
  const float* b_lepe = (const float*)d_in[5];
  const float* k_lepe = (const float*)d_in[6];
  const float* kb_lepe= (const float*)d_in[7];
  const float* w_sr   = (const float*)d_in[8];
  const float* b_sr   = (const float*)d_in[9];
  const float* w_q1   = (const float*)d_in[10];
  const float* w_kv1  = (const float*)d_in[11];
  const float* w_q2   = (const float*)d_in[12];
  const float* w_kv2  = (const float*)d_in[13];
  const float* w_proj = (const float*)d_in[14];
  const float* b_proj = (const float*)d_in[15];

  char* ws = (char*)d_ws;
  _Float16* xt16   = (_Float16*)(ws);              // 8 MB
  _Float16* xkt16  = (_Float16*)(ws + 8388608);    // 8 MB
  _Float16* q1q2   = (_Float16*)(ws + 16777216);   // 8 MB
  _Float16* lepe16 = (_Float16*)(ws + 25165824);   // 8 MB (NCHW)
  _Float16* lepc16 = (_Float16*)(ws + 33554432);   // 8 MB (NCHW)
  _Float16* kv2b   = (_Float16*)(ws + 41943040);   // 8 MB
  float*    x1     = (float*)   (ws + 50331648);   // 8 MB
  float*    wr     = (float*)   (ws + 58720256);   // 8 MB
  float*    xs     = (float*)   (ws + 67108864);   // 256 KB
  _Float16* k1s    = (_Float16*)(ws + 67371008);   // 64 KB
  _Float16* v1sT   = (_Float16*)(ws + 67436544);   // 64 KB
  _Float16* xsn16  = (_Float16*)(ws + 67502080);   // 128 KB
  float*    gmask  = (float*)   (ws + 67633152);   // 256 KB
  float*    lm     = (float*)   (ws + 67895296);   // 256 KB

  float* outp = (float*)d_out;

  k_zero     <<<256,   256, 0, stream>>>(lm);
  k_transpose<<<2048,  256, 0, stream>>>(x, xt16, xkt16);
  k_qlepe    <<<512,   256, 0, stream>>>(xt16, w_q1, w_q2, w_lepe, b_lepe, q1q2, lepe16);
  k_kv2      <<<512,   256, 0, stream>>>(xkt16, w_kv2, kv2b);
  k_lepeconv <<<16384, 256, 0, stream>>>(lepe16, k_lepe, kb_lepe, lepc16);
  k_srconv   <<<32,    256, 0, stream>>>(x, w_sr, b_sr, xs);
  k_kv1norm  <<<1,    1024, 0, stream>>>(xs, rho, gamma, beta, w_kv1, xsn16, k1s, v1sT);
  k_attn1    <<<256,   256, 0, stream>>>(q1q2, k1s, v1sT, x1, gmask);
  k_attn2    <<<512,   256, 0, stream>>>(q1q2, kv2b, wr, lm);
  k_final    <<<512,   256, 0, stream>>>(x1, wr, lepc16, w_proj, b_proj, outp);
  k_mask     <<<256,   256, 0, stream>>>(lm, gmask, outp + 4194304);
}
